// MyGCNModel_70506183131446
// MI455X (gfx1250) — matrix operand order repaired and checked by reading, not compiled, not run on hardware
//
#include <hip/hip_runtime.h>
#include <hip/hip_bf16.h>

// ---------------------------------------------------------------------------
// MI455X (gfx1250) GNN forward: GCN -> SAGE -> GAT -> GCN -> pool -> FC
// Dense GEMMs use v_wmma_f32_16x16x32_bf16 (bf16 in, f32 accumulate).
// Each wave computes a 16x64 output strip: A fragment loaded once per K-step,
// 4 independent WMMAs per step (no inter-WMMA hazards, clause-friendly loads).
// ---------------------------------------------------------------------------

#define NN 16384
#define EE 131072
#define GG 32
#define E2 (EE + NN)
#define EPS 1e-5f

typedef __attribute__((ext_vector_type(16))) __bf16 bf16x16;
typedef __attribute__((ext_vector_type(8)))  __bf16 bf16x8;
typedef __attribute__((ext_vector_type(8)))  float  f32x8;

// ---- helpers ---------------------------------------------------------------

__device__ __forceinline__ unsigned short f2bf(float f) {
    union { float f; unsigned int u; } v; v.f = f;
    unsigned int u = v.u;
    u += ((u >> 16) & 1u) + 0x7FFFu;   // round-to-nearest-even
    return (unsigned short)(u >> 16);
}

__device__ __forceinline__ void atomicMaxFloat(float* addr, float val) {
    unsigned int* ua = (unsigned int*)addr;
    unsigned int old = *ua;
    while (__uint_as_float(old) < val) {
        unsigned int assumed = old;
        old = atomicCAS(ua, assumed, __float_as_uint(val));
        if (old == assumed) break;
    }
}

// ---- generic small kernels -------------------------------------------------

__global__ void k_fill(float* p, float v, unsigned int n) {
    unsigned int g = blockIdx.x * 256u + threadIdx.x;
    if (g < n) p[g] = v;
}

__global__ void k_f32_to_bf16(const float* __restrict__ in, unsigned short* __restrict__ out,
                              unsigned int n) {
    unsigned int g = blockIdx.x * 256u + threadIdx.x;
    if (g < n) out[g] = f2bf(in[g]);
}

// w: [K,N] row-major f32  ->  wt: [N,K] row-major bf16
__global__ void k_transpose_bf16(const float* __restrict__ w, unsigned short* __restrict__ wt,
                                 int K, int N) {
    unsigned int g = blockIdx.x * 256u + threadIdx.x;
    if (g >= (unsigned int)(K * N)) return;
    int k = g / N, n = g % N;
    wt[(size_t)n * K + k] = f2bf(w[g]);
}

// ---- WMMA bf16 GEMM --------------------------------------------------------
// C[M,N] = A[M,K] @ Bt[N,K]^T (+bias[N]) (+Cin[M,N])
// One wave -> one 16x64 strip (4 adjacent 16x16 N-tiles).
// Requires: M % 16 == 0, N % 64 == 0, K % 32 == 0.

__global__ void k_gemm_bf16(const unsigned short* __restrict__ A,
                            const unsigned short* __restrict__ Bt,
                            const float* __restrict__ bias,
                            const float* __restrict__ Cin,
                            float* __restrict__ C,
                            int M, int N, int K) {
    int wave   = (int)(((size_t)blockIdx.x * blockDim.x + threadIdx.x) >> 5);
    int lane   = threadIdx.x & 31;
    int strips = N >> 6;                       // 64-wide strips
    int total  = (M >> 4) * strips;
    if (wave >= total) return;                 // whole-wave uniform: EXEC stays all-1s
    int tm = wave / strips, tn = wave % strips;
    int half = lane >> 4, row = lane & 15;

    const __bf16* a_base = (const __bf16*)A + (size_t)(tm * 16 + row) * K;
    // four B sub-tile bases (n-tile t covers columns tn*64 + t*16 .. +15)
    const __bf16* b_base[4];
#pragma unroll
    for (int t = 0; t < 4; ++t)
        b_base[t] = (const __bf16*)Bt + (size_t)(tn * 64 + t * 16 + row) * K + 16 * half;

    f32x8 c[4] = {{}, {}, {}, {}};
    if (Cin) {
#pragma unroll
        for (int t = 0; t < 4; ++t) {
            const float* cp = Cin + (size_t)(tm * 16 + 8 * half) * N + tn * 64 + t * 16 + row;
#pragma unroll
            for (int v = 0; v < 8; ++v) c[t][v] = cp[(size_t)v * N];
        }
    }
    if (bias) {
#pragma unroll
        for (int t = 0; t < 4; ++t) {
            float bv = bias[tn * 64 + t * 16 + row];
#pragma unroll
            for (int v = 0; v < 8; ++v) c[t][v] += bv;
        }
    }

    for (int k0 = 0; k0 < K; k0 += 32) {
        if (k0 + 32 < K) {                     // prefetch next K tile of A
            __builtin_prefetch(a_base + k0 + 32, 0, 3);
        }
        // A fragment 16x32: lane half 0 -> K {0..15}, half 1 -> K {16..31}
        bf16x8 alo = *(const bf16x8*)(a_base + k0 + 16 * half);
        bf16x8 ahi = *(const bf16x8*)(a_base + k0 + 16 * half + 8);
        bf16x16 a;
#pragma unroll
        for (int i = 0; i < 8; ++i) { a[i] = alo[i]; a[i + 8] = ahi[i]; }
        // 4 B fragments (contiguous 32B each) -> 4 independent WMMAs
        bf16x16 b0 = *(const bf16x16*)(b_base[0] + k0);
        bf16x16 b1 = *(const bf16x16*)(b_base[1] + k0);
        bf16x16 b2 = *(const bf16x16*)(b_base[2] + k0);
        bf16x16 b3 = *(const bf16x16*)(b_base[3] + k0);
        c[0] = __builtin_amdgcn_wmma_f32_16x16x32_bf16(false, a, false, b0, (short)0, c[0], false, false);
        c[1] = __builtin_amdgcn_wmma_f32_16x16x32_bf16(false, a, false, b1, (short)0, c[1], false, false);
        c[2] = __builtin_amdgcn_wmma_f32_16x16x32_bf16(false, a, false, b2, (short)0, c[2], false, false);
        c[3] = __builtin_amdgcn_wmma_f32_16x16x32_bf16(false, a, false, b3, (short)0, c[3], false, false);
    }

#pragma unroll
    for (int t = 0; t < 4; ++t) {
        float* op = C + (size_t)(tm * 16 + 8 * half) * N + tn * 64 + t * 16 + row;
#pragma unroll
        for (int v = 0; v < 8; ++v) op[(size_t)v * N] = c[t][v];
    }
}

// ---- graph kernels ---------------------------------------------------------

__global__ void k_edge_count(const int* __restrict__ d, float* cnt, unsigned int e) {
    unsigned int g = blockIdx.x * 256u + threadIdx.x;
    if (g < e) atomicAdd(&cnt[d[g]], 1.0f);
}

__global__ void k_dis(const float* __restrict__ deg, float* dis, unsigned int n) {
    unsigned int g = blockIdx.x * 256u + threadIdx.x;
    if (g < n) dis[g] = rsqrtf(deg[g] + 1.0f);   // +1 self loop
}

// gcn1 linear: h[N,64] = x[N,5] @ w[5,64]  (K too small for WMMA)
__global__ void k_gcn1(const float* __restrict__ x, const float* __restrict__ w,
                       float* __restrict__ h) {
    unsigned int g = blockIdx.x * 256u + threadIdx.x;
    if (g >= NN * 64u) return;
    int i = g >> 6, c = g & 63;
    float acc = 0.f;
#pragma unroll
    for (int j = 0; j < 5; ++j) acc += x[i * 5 + j] * w[j * 64 + c];
    h[g] = acc;
}

// out[i,c] = bias[c] + h[i,c]*dis[i]^2   (self-loop term of GCN)
__global__ void k_gcn_init(const float* __restrict__ h, const float* __restrict__ bias,
                           const float* __restrict__ dis, float* __restrict__ out,
                           unsigned int n, int C) {
    unsigned int g = blockIdx.x * 256u + threadIdx.x;
    if (g >= n * (unsigned int)C) return;
    int i = g / C, c = g % C;
    float d = dis[i];
    out[g] = bias[c] + h[g] * d * d;
}

__global__ void k_gcn_edges(const float* __restrict__ h, const int* __restrict__ src,
                            const int* __restrict__ dst, const float* __restrict__ dis,
                            float* __restrict__ out, int C) {
    unsigned int g = blockIdx.x * 256u + threadIdx.x;
    if (g >= (unsigned int)EE * (unsigned int)C) return;
    int e = g / C, c = g % C;
    int s = src[e], d = dst[e];
    atomicAdd(&out[(size_t)d * C + c], h[(size_t)s * C + c] * dis[s] * dis[d]);
}

// ---- batchnorm -------------------------------------------------------------

__global__ void k_bn_stats(const float* __restrict__ x, int n, int C,
                           float* mean, float* rstd) {
    int c = blockIdx.x;
    float s = 0.f, s2 = 0.f;
    for (int i = threadIdx.x; i < n; i += 256) {
        float v = x[(size_t)i * C + c];
        s += v; s2 += v * v;
    }
    __shared__ float sh[256], sh2[256];
    sh[threadIdx.x] = s; sh2[threadIdx.x] = s2;
    __syncthreads();
    for (int st = 128; st > 0; st >>= 1) {
        if (threadIdx.x < st) { sh[threadIdx.x] += sh[threadIdx.x + st];
                                sh2[threadIdx.x] += sh2[threadIdx.x + st]; }
        __syncthreads();
    }
    if (threadIdx.x == 0) {
        float mu = sh[0] / n;
        float var = sh2[0] / n - mu * mu;
        mean[c] = mu;
        rstd[c] = rsqrtf(var + EPS);
    }
}

__global__ void k_bn_relu(float* __restrict__ x, const float* __restrict__ mean,
                          const float* __restrict__ rstd, const float* __restrict__ g_,
                          const float* __restrict__ b_, unsigned int n, int C) {
    unsigned int g = blockIdx.x * 256u + threadIdx.x;
    if (g >= n * (unsigned int)C) return;
    int c = g % C;
    float v = (x[g] - mean[c]) * rstd[c] * g_[c] + b_[c];
    x[g] = fmaxf(v, 0.f);
}

// ---- SAGE ------------------------------------------------------------------

__global__ void k_sage_agg(const float* __restrict__ h, const int* __restrict__ src,
                           const int* __restrict__ dst, float* __restrict__ agg) {
    unsigned int g = blockIdx.x * 256u + threadIdx.x;
    if (g >= EE * 64u) return;
    int e = g >> 6, c = g & 63;
    atomicAdd(&agg[(size_t)dst[e] * 64 + c], h[(size_t)src[e] * 64 + c]);
}

__global__ void k_sage_scale(float* __restrict__ agg, const float* __restrict__ cnt) {
    unsigned int g = blockIdx.x * 256u + threadIdx.x;
    if (g >= NN * 64u) return;
    agg[g] /= fmaxf(cnt[g >> 6], 1.0f);
}

// ---- GAT -------------------------------------------------------------------

// es/ed: [N,4]
__global__ void k_attn_scores(const float* __restrict__ h3, const float* __restrict__ asrc,
                              const float* __restrict__ adst, float* __restrict__ es,
                              float* __restrict__ ed) {
    unsigned int g = blockIdx.x * 256u + threadIdx.x;
    if (g >= NN * 4u) return;
    int n = g >> 2, hd = g & 3;
    const float* hp = h3 + (size_t)n * 1024 + hd * 256;
    const float* as = asrc + hd * 256;
    const float* ad = adst + hd * 256;
    float s0 = 0.f, s1 = 0.f;
    for (int c = 0; c < 256; ++c) { s0 += hp[c] * as[c]; s1 += hp[c] * ad[c]; }
    es[g] = s0; ed[g] = s1;
}

__device__ __forceinline__ void gat_edge(unsigned int idx, const int* src, const int* dst,
                                         int& s, int& d) {
    if (idx < EE) { s = src[idx]; d = dst[idx]; }
    else          { s = (int)(idx - EE); d = s; }       // self loop
}

__global__ void k_gat_max(const float* __restrict__ es, const float* __restrict__ ed,
                          const int* __restrict__ src, const int* __restrict__ dst,
                          float* __restrict__ m) {
    unsigned int g = blockIdx.x * 256u + threadIdx.x;
    if (g >= (unsigned int)E2 * 4u) return;
    unsigned int idx = g >> 2; int hd = g & 3;
    int s, d; gat_edge(idx, src, dst, s, d);
    float e = es[s * 4 + hd] + ed[d * 4 + hd];
    e = (e > 0.f) ? e : 0.2f * e;                       // leaky_relu(0.2)
    atomicMaxFloat(&m[d * 4 + hd], e);
}

__global__ void k_gat_exp(const float* __restrict__ es, const float* __restrict__ ed,
                          const int* __restrict__ src, const int* __restrict__ dst,
                          const float* __restrict__ m, float* __restrict__ z,
                          float* __restrict__ alpha) {
    unsigned int g = blockIdx.x * 256u + threadIdx.x;
    if (g >= (unsigned int)E2 * 4u) return;
    unsigned int idx = g >> 2; int hd = g & 3;
    int s, d; gat_edge(idx, src, dst, s, d);
    float e = es[s * 4 + hd] + ed[d * 4 + hd];
    e = (e > 0.f) ? e : 0.2f * e;
    float ex = expf(e - m[d * 4 + hd]);
    alpha[g] = ex;
    atomicAdd(&z[d * 4 + hd], ex);
}

__global__ void k_gat_norm(float* __restrict__ alpha, const float* __restrict__ z,
                           const int* __restrict__ dst) {
    unsigned int g = blockIdx.x * 256u + threadIdx.x;
    if (g >= (unsigned int)E2 * 4u) return;
    unsigned int idx = g >> 2; int hd = g & 3;
    int d = (idx < EE) ? dst[idx] : (int)(idx - EE);
    alpha[g] /= z[d * 4 + hd];
}

__global__ void k_gat_scatter(const float* __restrict__ h3, const float* __restrict__ alpha,
                              const int* __restrict__ src, const int* __restrict__ dst,
                              float* __restrict__ gout) {
    unsigned int g = blockIdx.x * 256u + threadIdx.x;
    if (g >= (unsigned int)E2 * 1024u) return;
    unsigned int idx = g >> 10; int j = g & 1023; int hd = j >> 8;
    int s, d; gat_edge(idx, src, dst, s, d);
    atomicAdd(&gout[(size_t)d * 1024 + j], h3[(size_t)s * 1024 + j] * alpha[idx * 4 + hd]);
}

__global__ void k_head_mean(const float* __restrict__ gout, const float* __restrict__ bias,
                            float* __restrict__ out) {
    unsigned int g = blockIdx.x * 256u + threadIdx.x;
    if (g >= NN * 256u) return;
    int n = g >> 8, c = g & 255;
    const float* p = gout + (size_t)n * 1024;
    out[g] = 0.25f * (p[c] + p[256 + c] + p[512 + c] + p[768 + c]) + bias[c];
}

// ---- pooling ---------------------------------------------------------------

__global__ void k_batch_count(const int* __restrict__ batch, float* cnt) {
    unsigned int g = blockIdx.x * 256u + threadIdx.x;
    if (g < NN) atomicAdd(&cnt[batch[g]], 1.0f);
}

__global__ void k_pool(const float* __restrict__ h4, const int* __restrict__ batch,
                       float* __restrict__ psum, float* __restrict__ pmax) {
    unsigned int g = blockIdx.x * 256u + threadIdx.x;
    if (g >= NN * 512u) return;
    int i = g >> 9, c = g & 511;
    int gr = batch[i];
    float v = h4[g];
    atomicAdd(&psum[gr * 512 + c], v);
    atomicMaxFloat(&pmax[gr * 512 + c], v);
}

__global__ void k_pool_build(const float* __restrict__ psum, const float* __restrict__ pmax,
                             const float* __restrict__ cnt, float* __restrict__ pooled) {
    unsigned int g = blockIdx.x * 256u + threadIdx.x;
    if (g >= GG * 512u) return;
    int gr = g >> 9, c = g & 511;
    float cn = cnt[gr];
    pooled[gr * 1024 + c]       = psum[g] / fmaxf(cn, 1.0f);
    pooled[gr * 1024 + 512 + c] = (cn > 0.f) ? pmax[g] : 0.f;
}

// ---------------------------------------------------------------------------

static inline unsigned int nb(size_t n) { return (unsigned int)((n + 255) / 256); }

extern "C" void kernel_launch(void* const* d_in, const int* in_sizes, int n_in,
                              void* d_out, int out_size, void* d_ws, size_t ws_size,
                              hipStream_t stream) {
    (void)in_sizes; (void)n_in; (void)out_size; (void)ws_size;
    const float* x       = (const float*)d_in[0];
    const int*   ei      = (const int*)d_in[1];
    const int*   src     = ei;
    const int*   dst     = ei + EE;
    const int*   batch   = (const int*)d_in[2];
    const float* gcn1_w  = (const float*)d_in[3];
    const float* gcn1_b  = (const float*)d_in[4];
    const float* sage_wl = (const float*)d_in[5];
    const float* sage_wr = (const float*)d_in[6];
    const float* sage_b  = (const float*)d_in[7];
    const float* gat_w   = (const float*)d_in[8];
    const float* gat_as  = (const float*)d_in[9];
    const float* gat_ad  = (const float*)d_in[10];
    const float* gat_b   = (const float*)d_in[11];
    const float* gcn4_w  = (const float*)d_in[12];
    const float* gcn4_b  = (const float*)d_in[13];
    const float* bn1_g   = (const float*)d_in[14];
    const float* bn1_b   = (const float*)d_in[15];
    const float* bn2_g   = (const float*)d_in[16];
    const float* bn2_b   = (const float*)d_in[17];
    const float* bn3_g   = (const float*)d_in[18];
    const float* bn3_b   = (const float*)d_in[19];
    const float* bn4_g   = (const float*)d_in[20];
    const float* bn4_b   = (const float*)d_in[21];
    const float* fc_w    = (const float*)d_in[22];
    const float* fc_b    = (const float*)d_in[23];
    float* out = (float*)d_out;

    char* ws = (char*)d_ws;
    size_t off = 0;
    auto alloc = [&](size_t bytes) -> char* {
        char* p = ws + off;
        off += (bytes + 255) & ~(size_t)255;
        return p;
    };
    float* DEG   = (float*)alloc((size_t)NN * 4);
    float* DIS   = (float*)alloc((size_t)NN * 4);
    float* CNT   = (float*)alloc((size_t)NN * 4);
    float* MEANB = (float*)alloc(512 * 4);
    float* RSTDB = (float*)alloc(512 * 4);
    float* H1L   = (float*)alloc((size_t)NN * 64 * 4);
    float* H1    = (float*)alloc((size_t)NN * 64 * 4);
    float* AGG   = (float*)alloc((size_t)NN * 64 * 4);
    float* H2    = (float*)alloc((size_t)NN * 128 * 4);
    float* ES    = (float*)alloc((size_t)NN * 4 * 4);
    float* ED    = (float*)alloc((size_t)NN * 4 * 4);
    float* M4    = (float*)alloc((size_t)NN * 4 * 4);
    float* Z4    = (float*)alloc((size_t)NN * 4 * 4);
    float* ALPHA = (float*)alloc((size_t)E2 * 4 * 4);
    float* H3    = (float*)alloc((size_t)NN * 1024 * 4);
    float* GOUT  = (float*)alloc((size_t)NN * 1024 * 4);
    float* H3M   = (float*)alloc((size_t)NN * 256 * 4);
    float* H4L   = GOUT;                       // GOUT region free after head_mean
    float* H4    = GOUT + (size_t)NN * 512;
    float* PSUM  = (float*)alloc((size_t)GG * 512 * 4);
    float* PMAX  = (float*)alloc((size_t)GG * 512 * 4);
    float* CNTG  = (float*)alloc(GG * 4);
    float* POOLED= (float*)alloc((size_t)GG * 1024 * 4);
    unsigned short* ABF  = (unsigned short*)alloc((size_t)NN * 256 * 2);
    unsigned short* ABF2 = (unsigned short*)alloc((size_t)NN * 64 * 2);
    unsigned short* WBF  = (unsigned short*)alloc((size_t)1024 * 1024 * 2);

    const float NEG = -3.4e38f;

#define L(kern, n, ...) kern<<<nb(n), 256, 0, stream>>>(__VA_ARGS__)
    auto gemm = [&](const unsigned short* A, const unsigned short* Bt, const float* bias,
                    const float* Cin, float* C, int M, int N, int K) {
        size_t waves = (size_t)(M / 16) * (N / 64);   // one wave per 16x64 strip
        k_gemm_bf16<<<nb(waves * 32), 256, 0, stream>>>(A, Bt, bias, Cin, C, M, N, K);
    };

    // ---- degree / normalization (shared by gcn1 & gcn4) ----
    L(k_fill, NN, DEG, 0.f, NN);
    L(k_edge_count, EE, dst, DEG, EE);
    L(k_dis, NN, DEG, DIS, NN);

    // ---- layer 1: GCN(5->64) + BN + ReLU ----
    L(k_gcn1, (size_t)NN * 64, x, gcn1_w, H1L);
    L(k_gcn_init, (size_t)NN * 64, H1L, gcn1_b, DIS, H1, NN, 64);
    L(k_gcn_edges, (size_t)EE * 64, H1L, src, dst, DIS, H1, 64);
    k_bn_stats<<<64, 256, 0, stream>>>(H1, NN, 64, MEANB, RSTDB);
    L(k_bn_relu, (size_t)NN * 64, H1, MEANB, RSTDB, bn1_g, bn1_b, NN, 64);

    // ---- layer 2: SAGE(64->128) + BN + ReLU ----
    L(k_fill, NN, CNT, 0.f, NN);
    L(k_edge_count, EE, dst, CNT, EE);
    L(k_fill, (size_t)NN * 64, AGG, 0.f, NN * 64u);
    L(k_sage_agg, (size_t)EE * 64, H1, src, dst, AGG);
    L(k_sage_scale, (size_t)NN * 64, AGG, CNT);
    L(k_f32_to_bf16, (size_t)NN * 64, AGG, ABF, NN * 64u);
    L(k_f32_to_bf16, (size_t)NN * 64, H1, ABF2, NN * 64u);
    L(k_transpose_bf16, 64 * 128, sage_wl, WBF, 64, 128);
    gemm(ABF, WBF, sage_b, nullptr, H2, NN, 128, 64);
    L(k_transpose_bf16, 64 * 128, sage_wr, WBF, 64, 128);
    gemm(ABF2, WBF, nullptr, H2, H2, NN, 128, 64);
    k_bn_stats<<<128, 256, 0, stream>>>(H2, NN, 128, MEANB, RSTDB);
    L(k_bn_relu, (size_t)NN * 128, H2, MEANB, RSTDB, bn2_g, bn2_b, NN, 128);

    // ---- layer 3: GAT(128->4x256) + BN + ReLU ----
    L(k_f32_to_bf16, (size_t)NN * 128, H2, ABF, NN * 128u);
    L(k_transpose_bf16, 128 * 1024, gat_w, WBF, 128, 1024);
    gemm(ABF, WBF, nullptr, nullptr, H3, NN, 1024, 128);
    L(k_attn_scores, (size_t)NN * 4, H3, gat_as, gat_ad, ES, ED);
    L(k_fill, (size_t)NN * 4, M4, NEG, NN * 4u);
    L(k_gat_max, (size_t)E2 * 4, ES, ED, src, dst, M4);
    L(k_fill, (size_t)NN * 4, Z4, 0.f, NN * 4u);
    L(k_gat_exp, (size_t)E2 * 4, ES, ED, src, dst, M4, Z4, ALPHA);
    L(k_gat_norm, (size_t)E2 * 4, ALPHA, Z4, dst);
    L(k_fill, (size_t)NN * 1024, GOUT, 0.f, NN * 1024u);
    L(k_gat_scatter, (size_t)E2 * 1024, H3, ALPHA, src, dst, GOUT);
    L(k_head_mean, (size_t)NN * 256, GOUT, gat_b, H3M);
    k_bn_stats<<<256, 256, 0, stream>>>(H3M, NN, 256, MEANB, RSTDB);
    L(k_bn_relu, (size_t)NN * 256, H3M, MEANB, RSTDB, bn3_g, bn3_b, NN, 256);

    // ---- layer 4: GCN(256->512) + BN + ReLU ----
    L(k_f32_to_bf16, (size_t)NN * 256, H3M, ABF, NN * 256u);
    L(k_transpose_bf16, 256 * 512, gcn4_w, WBF, 256, 512);
    gemm(ABF, WBF, nullptr, nullptr, H4L, NN, 512, 256);
    L(k_gcn_init, (size_t)NN * 512, H4L, gcn4_b, DIS, H4, NN, 512);
    L(k_gcn_edges, (size_t)EE * 512, H4L, src, dst, DIS, H4, 512);
    k_bn_stats<<<512, 256, 0, stream>>>(H4, NN, 512, MEANB, RSTDB);
    L(k_bn_relu, (size_t)NN * 512, H4, MEANB, RSTDB, bn4_g, bn4_b, NN, 512);

    // ---- pooling ----
    L(k_fill, GG, CNTG, 0.f, GG);
    L(k_batch_count, NN, batch, CNTG);
    L(k_fill, (size_t)GG * 512, PSUM, 0.f, GG * 512u);
    L(k_fill, (size_t)GG * 512, PMAX, NEG, GG * 512u);
    L(k_pool, (size_t)NN * 512, H4, batch, PSUM, PMAX);
    L(k_pool_build, (size_t)GG * 512, PSUM, PMAX, CNTG, POOLED);

    // ---- FC: [32,1024] @ [1024,1024] + b -> d_out ----
    L(k_f32_to_bf16, (size_t)GG * 1024, POOLED, ABF, GG * 1024u);
    L(k_transpose_bf16, 1024 * 1024, fc_w, WBF, 1024, 1024);
    gemm(ABF, WBF, fc_b, nullptr, out, GG, 1024, 1024);
#undef L
}
